// MultiQueryAttention_16887811408073
// MI455X (gfx1250) — matrix operand order repaired and checked by reading, not compiled, not run on hardware
//
#include <hip/hip_runtime.h>
#include <hip/hip_bf16.h>
#include <stdint.h>

typedef __attribute__((ext_vector_type(16))) __bf16 v16bf;
typedef __attribute__((ext_vector_type(8)))  float  v8f;

typedef unsigned short u16;
typedef unsigned int   u32;

#define B_  2
#define T_  2048
#define D_  2048
#define H_  16
#define HD_ 128
#define BT_ (B_ * T_)

__device__ __forceinline__ u16 f32_to_bf16(float f) {
  u32 u = __float_as_uint(f);
  u32 r = u + 0x7FFFu + ((u >> 16) & 1u);   // round-to-nearest-even
  return (u16)(r >> 16);
}

union Frag16 {
  v16bf v;
  u16   u[16];
  u32   d[8];
  uint4 q[2];
};

// ---------------------------------------------------------------- cast
__global__ __launch_bounds__(256) void cast_f32_bf16(const float* __restrict__ in,
                                                     u16* __restrict__ out, int n) {
  int i = blockIdx.x * 256 + threadIdx.x;
  if (i < n) out[i] = f32_to_bf16(in[i]);
}

// ---------------------------------------------------------------- WMMA GEMM
// C (MxN f32, row-major) = A (MxK bf16, row-major) x B (KxN bf16, row-major)
// block tile: 128(M) x 128(N), K-step 32, double-buffered LDS, one barrier/step.
__global__ __launch_bounds__(256) void gemm_bf16_f32(const u16* __restrict__ A,
                                                     const u16* __restrict__ Bm,
                                                     float* __restrict__ C,
                                                     int M, int N, int K) {
  __shared__ __align__(16) u16 sA[2][128 * 32];    // [buf][row][k]
  __shared__ __align__(16) u16 sBt[2][128 * 32];   // [buf][col][k] (transposed)
  const int tid = threadIdx.x;
  const int lane = tid & 31;
  const int wave = tid >> 5;
  const int ln = lane & 15;
  const int halfsel = lane >> 4;          // 0: lanes 0-15, 1: lanes 16-31
  const int mbase = blockIdx.y * 128;
  const int nbase = blockIdx.x * 128;

  const int ar = tid >> 1, acb = (tid & 1) * 16;   // A staging: row, col-base
  const int br = tid >> 3, bcb = (tid & 7) * 16;   // B staging: k-row, col-base

  uint4 ga0, ga1, gb0, gb1;                        // in-flight global tile

  auto loadG = [&](int k0) {
    const uint4* g = reinterpret_cast<const uint4*>(A + (size_t)(mbase + ar) * K + k0 + acb);
    ga0 = g[0];
    ga1 = g[1];
    const u16* gB = Bm + (size_t)(k0 + br) * N + nbase + bcb;
    gb0 = *reinterpret_cast<const uint4*>(gB);
    gb1 = *reinterpret_cast<const uint4*>(gB + 8);
  };
  auto storeS = [&](int buf) {
    uint4* s = reinterpret_cast<uint4*>(&sA[buf][ar * 32 + acb]);
    s[0] = ga0;
    s[1] = ga1;
    const u16* pv = reinterpret_cast<const u16*>(&gb0);
#pragma unroll
    for (int e = 0; e < 8; ++e) sBt[buf][(bcb + e) * 32 + br] = pv[e];
    pv = reinterpret_cast<const u16*>(&gb1);
#pragma unroll
    for (int e = 0; e < 8; ++e) sBt[buf][(bcb + 8 + e) * 32 + br] = pv[e];
  };

  v8f acc[8] = {};
  const int nk = K >> 5;

  loadG(0);
  storeS(0);
  __syncthreads();

  for (int t = 0; t < nk; ++t) {
    const int buf = t & 1;
    if (t + 1 < nk) loadG((t + 1) * 32);   // issue next tile's HBM loads early

    // prefetch all fragments, then a clean 8-deep WMMA chain
    Frag16 a;
    {
      const int arow = wave * 16 + ln;
      const int kb0 = halfsel * 16;
      a.q[0] = *reinterpret_cast<const uint4*>(&sA[buf][arow * 32 + kb0]);
      a.q[1] = *reinterpret_cast<const uint4*>(&sA[buf][arow * 32 + kb0 + 8]);
    }
    Frag16 b[8];
#pragma unroll
    for (int c = 0; c < 8; ++c) {
      const uint4* bp = reinterpret_cast<const uint4*>(&sBt[buf][(c * 16 + ln) * 32 + halfsel * 16]);
      b[c].q[0] = bp[0];
      b[c].q[1] = bp[1];
    }
#pragma unroll
    for (int c = 0; c < 8; ++c)
      acc[c] = __builtin_amdgcn_wmma_f32_16x16x32_bf16(false, a.v, false, b[c].v,
                                                       (short)0, acc[c], false, false);

    if (t + 1 < nk) storeS(buf ^ 1);       // LDS stores land behind the compute
    __syncthreads();
  }

  // writeback: vgpr i -> row i (+8 upper half); lane&15 -> col
  {
    int row0 = mbase + wave * 16 + halfsel * 8;
    int colb = nbase + ln;
#pragma unroll
    for (int c = 0; c < 8; ++c) {
#pragma unroll
      for (int i = 0; i < 8; ++i)
        C[(size_t)(row0 + i) * N + colb + c * 16] = acc[c][i];
    }
  }
}

// ---------------------------------------------------------------- RoPE q -> [B,H,T,128] bf16
__global__ __launch_bounds__(256) void rope_pack_q(const float* __restrict__ qf,
                                                   u16* __restrict__ qb) {
  int p = blockIdx.x * 256 + threadIdx.x;        // B*T*H*64 = 2^22
  if (p >= B_ * T_ * H_ * 64) return;
  int i = p & 63;
  int h = (p >> 6) & (H_ - 1);
  int t = (p >> 10) & (T_ - 1);
  int b = p >> 21;
  const float* src = qf + ((size_t)(b * T_ + t)) * D_ + h * HD_;
  float x1 = src[i], x2 = src[i + 64];
  float e = (float)(2 * i) * (1.0f / 128.0f);
  float invf = __powf(10000.0f, -e);
  float ang = (float)t * invf;
  float cs, sn;
  __sincosf(ang, &cs, &sn);
  float o1 = x1 * cs - x2 * sn;
  float o2 = x2 * cs + x1 * sn;
  u16* dst = qb + (((size_t)(b * H_ + h)) * T_ + t) * HD_;
  dst[i] = f32_to_bf16(o1);
  dst[i + 64] = f32_to_bf16(o2);
}

// ---------------------------------------------------------------- RoPE k -> bf16 + f32 broadcast out
__global__ __launch_bounds__(256) void rope_pack_k(const float* __restrict__ kf,
                                                   u16* __restrict__ kb,
                                                   float* __restrict__ kexp) {
  int p = blockIdx.x * 256 + threadIdx.x;        // B*T*64
  if (p >= B_ * T_ * 64) return;
  int i = p & 63;
  int t = (p >> 6) & (T_ - 1);
  int b = p >> 17;
  const float* src = kf + ((size_t)(b * T_ + t)) * HD_;
  float x1 = src[i], x2 = src[i + 64];
  float e = (float)(2 * i) * (1.0f / 128.0f);
  float invf = __powf(10000.0f, -e);
  float ang = (float)t * invf;
  float cs, sn;
  __sincosf(ang, &cs, &sn);
  float o1 = x1 * cs - x2 * sn;
  float o2 = x2 * cs + x1 * sn;
  u16* dst = kb + ((size_t)(b * T_ + t)) * HD_;
  dst[i] = f32_to_bf16(o1);
  dst[i + 64] = f32_to_bf16(o2);
#pragma unroll
  for (int h = 0; h < H_; ++h) {
    float* o = kexp + (((size_t)(b * H_ + h)) * T_ + t) * HD_;
    o[i] = o1;
    o[i + 64] = o2;
  }
}

// ---------------------------------------------------------------- v -> bf16 + f32 broadcast out
__global__ __launch_bounds__(256) void pack_v(const float* __restrict__ vf,
                                              u16* __restrict__ vb,
                                              float* __restrict__ vexp) {
  int p = blockIdx.x * 256 + threadIdx.x;        // B*T*128
  if (p >= B_ * T_ * HD_) return;
  int d = p & (HD_ - 1);
  int t = (p >> 7) & (T_ - 1);
  int b = p >> 18;
  float v = vf[((size_t)(b * T_ + t)) * HD_ + d];
  vb[((size_t)(b * T_ + t)) * HD_ + d] = f32_to_bf16(v);
#pragma unroll
  for (int h = 0; h < H_; ++h)
    vexp[(((size_t)(b * H_ + h)) * T_ + t) * HD_ + d] = v;
}

// ---------------------------------------------------------------- flash MQA attention
// grid: (T/128, B*H). Each wave: 16 query rows, streaming softmax over causal
// 32-key tiles, double-buffered K/V staging, batched WMMA fragment prefetch.
__global__ __launch_bounds__(256) void mqa_attn(const u16* __restrict__ Q,   // [B*H][T][128] bf16
                                                const u16* __restrict__ Kb,  // [B][T][128] bf16
                                                const u16* __restrict__ Vb,  // [B][T][128] bf16
                                                u16* __restrict__ Y) {       // [B][T][H*128] bf16
  __shared__ __align__(16) u16 sK[2][32 * 128];    // [buf][key][hd]
  __shared__ __align__(16) u16 sVt[2][128 * 32];   // [buf][hd col][key]
  __shared__ __align__(16) u16 sP[8][16 * 32];     // per-wave P scratch

  const int tid = threadIdx.x;
  const int lane = tid & 31;
  const int wave = tid >> 5;
  const int ln = lane & 15;
  const int halfsel = lane >> 4;
  const int bh = blockIdx.y;
  const int b = bh >> 4;
  const int h = bh & (H_ - 1);
  const int R0 = blockIdx.x * 128;
  const int r0 = R0 + wave * 16;

  const int kr = tid >> 3;                 // staged key row 0..31
  const int kcb = (tid & 7) * 16;          // staged hd base
  uint4 gk0, gk1, gv0, gv1;                // in-flight K/V tile

  auto loadKV = [&](int kb0) {
    const u16* gk = Kb + ((size_t)(b * T_) + kb0 + kr) * HD_ + kcb;
    gk0 = *reinterpret_cast<const uint4*>(gk);
    gk1 = *reinterpret_cast<const uint4*>(gk + 8);
    const u16* gv = Vb + ((size_t)(b * T_) + kb0 + kr) * HD_ + kcb;
    gv0 = *reinterpret_cast<const uint4*>(gv);
    gv1 = *reinterpret_cast<const uint4*>(gv + 8);
  };
  auto storeKV = [&](int buf) {
    *reinterpret_cast<uint4*>(&sK[buf][kr * 128 + kcb]) = gk0;
    *reinterpret_cast<uint4*>(&sK[buf][kr * 128 + kcb + 8]) = gk1;
    const u16* pv = reinterpret_cast<const u16*>(&gv0);
#pragma unroll
    for (int e = 0; e < 8; ++e) sVt[buf][(kcb + e) * 32 + kr] = pv[e];
    pv = reinterpret_cast<const u16*>(&gv1);
#pragma unroll
    for (int e = 0; e < 8; ++e) sVt[buf][(kcb + 8 + e) * 32 + kr] = pv[e];
  };

  // resident Q fragments (4 K-steps of 32 over head_dim=128)
  Frag16 qf[4];
  {
    const u16* qrow = Q + (((size_t)bh) * T_ + (r0 + ln)) * HD_;
#pragma unroll
    for (int f = 0; f < 4; ++f) {
      int hd0 = 32 * f + halfsel * 16;
      qf[f].q[0] = *reinterpret_cast<const uint4*>(qrow + hd0);
      qf[f].q[1] = *reinterpret_cast<const uint4*>(qrow + hd0 + 8);
    }
  }

  v8f acc[8] = {};
  float mrow[8], lrow[8];
#pragma unroll
  for (int i = 0; i < 8; ++i) { mrow[i] = -1e30f; lrow[i] = 0.0f; }
  const float sc = 0.08838834764831845f;   // 1/sqrt(128)

  const int ktiles = (R0 >> 5) + 4;        // keys [0, R0+128)

  loadKV(0);
  storeKV(0);
  __syncthreads();

  for (int kt = 0; kt < ktiles; ++kt) {
    const int kb = kt * 32;
    const int buf = kt & 1;
    if (kt + 1 < ktiles) loadKV((kt + 1) * 32);   // next tile's HBM loads in flight

    if (kb <= r0 + 15) {
      // ---- S = Q K^T : prefetch all 8 K fragments, then 8 WMMAs
      Frag16 bk[8];
#pragma unroll
      for (int f = 0; f < 4; ++f) {
        const uint4* bp0 = reinterpret_cast<const uint4*>(&sK[buf][ln * 128 + 32 * f + halfsel * 16]);
        bk[f].q[0] = bp0[0];
        bk[f].q[1] = bp0[1];
        const uint4* bp1 = reinterpret_cast<const uint4*>(&sK[buf][(16 + ln) * 128 + 32 * f + halfsel * 16]);
        bk[4 + f].q[0] = bp1[0];
        bk[4 + f].q[1] = bp1[1];
      }
      v8f s0 = {}, s1 = {};
#pragma unroll
      for (int f = 0; f < 4; ++f)
        s0 = __builtin_amdgcn_wmma_f32_16x16x32_bf16(false, qf[f].v, false, bk[f].v,
                                                     (short)0, s0, false, false);
#pragma unroll
      for (int f = 0; f < 4; ++f)
        s1 = __builtin_amdgcn_wmma_f32_16x16x32_bf16(false, qf[f].v, false, bk[4 + f].v,
                                                     (short)0, s1, false, false);

      // ---- streaming softmax update (each row lives in one 16-lane half)
      float p0[8], p1[8], rscale[8];
#pragma unroll
      for (int i = 0; i < 8; ++i) {
        int row = r0 + i + halfsel * 8;
        float x0 = s0[i] * sc;
        float x1 = s1[i] * sc;
        if (kb + ln > row)      x0 = -1e30f;   // causal mask
        if (kb + 16 + ln > row) x1 = -1e30f;
        float t = fmaxf(x0, x1);
        t = fmaxf(t, __shfl_xor(t, 1, 32));
        t = fmaxf(t, __shfl_xor(t, 2, 32));
        t = fmaxf(t, __shfl_xor(t, 4, 32));
        t = fmaxf(t, __shfl_xor(t, 8, 32));
        float nm  = fmaxf(mrow[i], t);
        float sc0 = __expf(mrow[i] - nm);
        float e0  = __expf(x0 - nm);
        float e1  = __expf(x1 - nm);
        float rs  = e0 + e1;
        rs += __shfl_xor(rs, 1, 32);
        rs += __shfl_xor(rs, 2, 32);
        rs += __shfl_xor(rs, 4, 32);
        rs += __shfl_xor(rs, 8, 32);
        lrow[i] = lrow[i] * sc0 + rs;
        mrow[i] = nm;
        rscale[i] = sc0;
        p0[i] = e0;
        p1[i] = e1;
      }
#pragma unroll
      for (int c = 0; c < 8; ++c) {
#pragma unroll
        for (int i = 0; i < 8; ++i) acc[c][i] *= rscale[i];
      }

      // ---- P to per-wave LDS (accumulator layout -> A-fragment layout)
#pragma unroll
      for (int i = 0; i < 8; ++i) {
        int rl = i + halfsel * 8;
        sP[wave][rl * 32 + ln]      = f32_to_bf16(p0[i]);
        sP[wave][rl * 32 + 16 + ln] = f32_to_bf16(p1[i]);
      }
      Frag16 pa;
      pa.q[0] = *reinterpret_cast<const uint4*>(&sP[wave][ln * 32 + halfsel * 16]);
      pa.q[1] = *reinterpret_cast<const uint4*>(&sP[wave][ln * 32 + halfsel * 16 + 8]);

      // ---- O += P (16x32) x V (32x128): prefetch all 8 V fragments, then 8 WMMAs
      Frag16 bv[8];
#pragma unroll
      for (int c = 0; c < 8; ++c) {
        const uint4* bp = reinterpret_cast<const uint4*>(&sVt[buf][(c * 16 + ln) * 32 + halfsel * 16]);
        bv[c].q[0] = bp[0];
        bv[c].q[1] = bp[1];
      }
#pragma unroll
      for (int c = 0; c < 8; ++c)
        acc[c] = __builtin_amdgcn_wmma_f32_16x16x32_bf16(false, pa.v, false, bv[c].v,
                                                         (short)0, acc[c], false, false);
    }

    if (kt + 1 < ktiles) storeKV(buf ^ 1);
    __syncthreads();
  }

  // normalize and write y as bf16 [B][T][H*128]
  {
    int row = r0 + halfsel * 8;
#pragma unroll
    for (int i = 0; i < 8; ++i) {
      float inv = 1.0f / lrow[i];
      size_t base = ((size_t)(b * T_) + row + i) * D_ + h * HD_ + ln;
#pragma unroll
      for (int c = 0; c < 8; ++c)
        Y[base + c * 16] = f32_to_bf16(acc[c][i] * inv);
    }
  }
}

// ---------------------------------------------------------------- launcher
extern "C" void kernel_launch(void* const* d_in, const int* in_sizes, int n_in,
                              void* d_out, int out_size, void* d_ws, size_t ws_size,
                              hipStream_t stream) {
  const float* x  = (const float*)d_in[0];
  // d_in[1] = attn_mask: exactly causal -1e9 above diagonal; applied analytically.
  const float* wq = (const float*)d_in[2];
  const float* wk = (const float*)d_in[3];
  const float* wv = (const float*)d_in[4];
  const float* wo = (const float*)d_in[5];

  char* w = (char*)d_ws;
  size_t off = 0;
  auto alloc = [&](size_t bytes) -> void* {
    void* p = w + off;
    off = (off + bytes + 255) & ~(size_t)255;
    return p;
  };

  u16* xb    = (u16*)alloc((size_t)BT_ * D_ * 2);
  u16* wqb   = (u16*)alloc((size_t)D_ * D_ * 2);
  u16* wkb   = (u16*)alloc((size_t)D_ * HD_ * 2);
  u16* wvb   = (u16*)alloc((size_t)D_ * HD_ * 2);
  u16* wob   = (u16*)alloc((size_t)D_ * D_ * 2);
  float* qfb = (float*)alloc((size_t)BT_ * D_ * 4);
  float* kfb = (float*)alloc((size_t)BT_ * HD_ * 4);
  float* vfb = (float*)alloc((size_t)BT_ * HD_ * 4);
  u16* qb    = (u16*)alloc((size_t)B_ * H_ * T_ * HD_ * 2);
  u16* kb    = (u16*)alloc((size_t)BT_ * HD_ * 2);
  u16* vb    = (u16*)alloc((size_t)BT_ * HD_ * 2);
  u16* yb    = (u16*)alloc((size_t)BT_ * D_ * 2);

  float* out_y = (float*)d_out;
  float* out_k = out_y + (size_t)BT_ * D_;
  float* out_v = out_k + (size_t)B_ * H_ * T_ * HD_;

  auto g1 = [](int n) { return dim3((unsigned)((n + 255) / 256)); };

  cast_f32_bf16<<<g1(BT_ * D_), 256, 0, stream>>>(x, xb, BT_ * D_);
  cast_f32_bf16<<<g1(D_ * D_), 256, 0, stream>>>(wq, wqb, D_ * D_);
  cast_f32_bf16<<<g1(D_ * HD_), 256, 0, stream>>>(wk, wkb, D_ * HD_);
  cast_f32_bf16<<<g1(D_ * HD_), 256, 0, stream>>>(wv, wvb, D_ * HD_);
  cast_f32_bf16<<<g1(D_ * D_), 256, 0, stream>>>(wo, wob, D_ * D_);

  gemm_bf16_f32<<<dim3(D_ / 128, BT_ / 128), 256, 0, stream>>>(xb, wqb, qfb, BT_, D_, D_);
  gemm_bf16_f32<<<dim3(HD_ / 128, BT_ / 128), 256, 0, stream>>>(xb, wkb, kfb, BT_, HD_, D_);
  gemm_bf16_f32<<<dim3(HD_ / 128, BT_ / 128), 256, 0, stream>>>(xb, wvb, vfb, BT_, HD_, D_);

  rope_pack_q<<<g1(B_ * T_ * H_ * 64), 256, 0, stream>>>(qfb, qb);
  rope_pack_k<<<g1(B_ * T_ * 64), 256, 0, stream>>>(kfb, kb, out_k);
  pack_v<<<g1(B_ * T_ * HD_), 256, 0, stream>>>(vfb, vb, out_v);

  mqa_attn<<<dim3(T_ / 128, B_ * H_), 256, 0, stream>>>(qb, kb, vb, yb);

  gemm_bf16_f32<<<dim3(D_ / 128, BT_ / 128), 256, 0, stream>>>(yb, wob, out_y, BT_, D_, D_);
}
